// EGNNConv_12515534701202
// MI455X (gfx1250) — matrix operand order repaired and checked by reading, not compiled, not run on hardware
//
#include <hip/hip_runtime.h>
#include <hip/hip_bf16.h>

// ---------------------------------------------------------------------------
// EGNN layer for MI455X (gfx1250): f16 WMMA (16x16x32) with fp32 accumulate.
// h is pre-converted to f16 once (workspace permitting) so the 32x-redundant
// per-edge gathers move half the bytes and need no per-tile conversion VALU.
// ---------------------------------------------------------------------------

typedef _Float16 f16;
typedef __attribute__((ext_vector_type(16))) _Float16 v16h;
typedef __attribute__((ext_vector_type(8)))  _Float16 v8h;
typedef __attribute__((ext_vector_type(8)))  float    v8f;

#define WAVES_PER_BLOCK 4
#define BLOCK_THREADS (WAVES_PER_BLOCK * 32)

__device__ __forceinline__ float silu_f(float x) {
    return x * (1.0f / (1.0f + __expf(-x)));
}

// Build a 16-element f16 fragment from two 16-byte LDS chunks.
__device__ __forceinline__ v16h make_frag(const f16* lo, const f16* hi) {
    v8h a = *(const v8h*)lo;
    v8h b = *(const v8h*)hi;
    return __builtin_shufflevector(a, b, 0,1,2,3,4,5,6,7,8,9,10,11,12,13,14,15);
}

// ---------------------------------------------------------------------------
// Kernel 0: zero msg_agg, pos_out = pos, and (optionally) h16 = (f16)h.
// ---------------------------------------------------------------------------
__global__ void egnn_init_kernel(const float* __restrict__ h,
                                 const float* __restrict__ pos,
                                 float* __restrict__ msgAgg,
                                 float* __restrict__ posOut,
                                 f16* __restrict__ h16,   // may be null
                                 int N) {
    int i = blockIdx.x * blockDim.x + threadIdx.x;
    if (i < N * 64) {
        msgAgg[i] = 0.0f;
        if (h16) h16[i] = (f16)h[i];
    }
    if (i < N * 3) posOut[i] = pos[i];
}

// ---------------------------------------------------------------------------
// Kernel 1: per-edge MLPs (edge_mlp + coord head) + atomic scatter.
// One wave handles a tile of 16 edges.
// ---------------------------------------------------------------------------
template <bool F16H>
__global__ __launch_bounds__(BLOCK_THREADS)
void egnn_edge_kernel(const float* __restrict__ h,
                      const f16* __restrict__ h16,
                      const float* __restrict__ pos,
                      const long long* __restrict__ eidx,
                      const float* __restrict__ We1, const float* __restrict__ be1,
                      const float* __restrict__ We2, const float* __restrict__ be2,
                      const float* __restrict__ Wc1, const float* __restrict__ bc1,
                      const float* __restrict__ Wc2, const float* __restrict__ bc2,
                      float* __restrict__ msgAgg,   // [N,64] fp32 accumulator
                      float* __restrict__ posOut,   // [N,3]  fp32 accumulator
                      int N, int E)
{
    // Weights transposed into LDS as f16: sWT[n][k]. K of layer 1 padded 129->160.
    __shared__ __align__(16) f16 sWT1[64 * 160];
    __shared__ __align__(16) f16 sWT2[64 * 64];
    __shared__ __align__(16) f16 sWTc[64 * 64];
    __shared__ float sBe1[64], sBe2[64], sBc1[64], sWc2v[64];
    __shared__ __align__(16) f16 sStage[WAVES_PER_BLOCK][16 * 160];
    __shared__ int   sRow[WAVES_PER_BLOCK][16];
    __shared__ float sRel[WAVES_PER_BLOCK][16][3];
    __shared__ float sCw[WAVES_PER_BLOCK][16];

    const int tid = threadIdx.x;
    for (int i = tid; i < 64 * 160; i += BLOCK_THREADS) sWT1[i] = (f16)0.0f;
    __syncthreads();
    for (int i = tid; i < 129 * 64; i += BLOCK_THREADS) {
        int k = i >> 6, n = i & 63;
        sWT1[n * 160 + k] = (f16)We1[i];
    }
    for (int i = tid; i < 64 * 64; i += BLOCK_THREADS) {
        int k = i >> 6, n = i & 63;
        sWT2[n * 64 + k] = (f16)We2[i];
        sWTc[n * 64 + k] = (f16)Wc1[i];
    }
    if (tid < 64) {
        sBe1[tid]  = be1[tid];
        sBe2[tid]  = be2[tid];
        sBc1[tid]  = bc1[tid];
        sWc2v[tid] = Wc2[tid];
    }
    __syncthreads();

    const float bc2v = bc2[0];
    const int lane = tid & 31;
    const int wid  = tid >> 5;
    const int half = lane >> 4;   // 0: lanes 0-15, 1: lanes 16-31
    const int lm   = lane & 15;
    const int ntiles = E >> 4;    // E is a multiple of 16

    f16* st = sStage[wid];

    // Zero-pad K columns 129..159 of this wave's stage once (rows 0..5 of it get overwritten
    // by mid-stage reuse at flat idx < 1024; gathers touch cols 0..128).
    if (half == 0) {
        #pragma unroll
        for (int k2 = 129; k2 < 160; ++k2) st[lm * 160 + k2] = (f16)0.0f;
    }

    for (int tile = blockIdx.x * WAVES_PER_BLOCK + wid; tile < ntiles;
         tile += gridDim.x * WAVES_PER_BLOCK)
    {
        const int e = (tile << 4) + lm;
        const long long r = eidx[e];
        const long long c = eidx[E + e];
        const long long src = half ? c : r;

        if (half == 0) {
            sRow[wid][lm] = (int)r;
            float rx = pos[r * 3 + 0] - pos[c * 3 + 0];
            float ry = pos[r * 3 + 1] - pos[c * 3 + 1];
            float rz = pos[r * 3 + 2] - pos[c * 3 + 2];
            sRel[wid][lm][0] = rx;
            sRel[wid][lm][1] = ry;
            sRel[wid][lm][2] = rz;
            st[lm * 160 + 128] = (f16)(rx * rx + ry * ry + rz * rz);
        }
        // Gather 64 features of h[row] (lanes 0-15) / h[col] (lanes 16-31).
        if (F16H) {
            const uint4* hp = (const uint4*)(h16 + src * 64);
            uint4* dst = (uint4*)(st + lm * 160 + half * 64);
            #pragma unroll
            for (int k4 = 0; k4 < 8; ++k4) dst[k4] = hp[k4];
        } else {
            const float4* hp = (const float4*)(h + src * 64);
            f16* dst = st + lm * 160 + half * 64;
            #pragma unroll
            for (int k4 = 0; k4 < 16; ++k4) {
                float4 v = hp[k4];
                dst[k4 * 4 + 0] = (f16)v.x;
                dst[k4 * 4 + 1] = (f16)v.y;
                dst[k4 * 4 + 2] = (f16)v.z;
                dst[k4 * 4 + 3] = (f16)v.w;
            }
        }

        // ---- edge_mlp layer 1: [16 x 160] @ [160 x 64] ----
        v16h A[5];
        #pragma unroll
        for (int kc = 0; kc < 5; ++kc)
            A[kc] = make_frag(&st[lm * 160 + kc * 32 + half * 8],
                              &st[lm * 160 + kc * 32 + half * 8 + 16]);
        v8f C1[4];
        #pragma unroll
        for (int t = 0; t < 4; ++t) {
            v8f acc = {};
            const int n = lm + 16 * t;
            #pragma unroll
            for (int kc = 0; kc < 5; ++kc) {
                v16h B = make_frag(&sWT1[n * 160 + kc * 32 + half * 8],
                                   &sWT1[n * 160 + kc * 32 + half * 8 + 16]);
                acc = __builtin_amdgcn_wmma_f32_16x16x32_f16(
                          false, A[kc], false, B, (short)0, acc, false, false);
            }
            C1[t] = acc;
        }
        // bias + SiLU -> mid stage (C-layout -> A-layout via LDS), stride 64.
        #pragma unroll
        for (int t = 0; t < 4; ++t) {
            const int n = lm + 16 * t;
            const float bias = sBe1[n];
            #pragma unroll
            for (int v = 0; v < 8; ++v)
                st[(v + 8 * half) * 64 + n] = (f16)silu_f(C1[t][v] + bias);
        }

        // ---- edge_mlp layer 2: [16 x 64] @ [64 x 64] ----
        v16h A2[2];
        #pragma unroll
        for (int kc = 0; kc < 2; ++kc)
            A2[kc] = make_frag(&st[lm * 64 + kc * 32 + half * 8],
                               &st[lm * 64 + kc * 32 + half * 8 + 16]);
        v8f M[4];  // edge_msg, fp32, kept for scatter
        #pragma unroll
        for (int t = 0; t < 4; ++t) {
            v8f acc = {};
            const int n = lm + 16 * t;
            #pragma unroll
            for (int kc = 0; kc < 2; ++kc) {
                v16h B = make_frag(&sWT2[n * 64 + kc * 32 + half * 8],
                                   &sWT2[n * 64 + kc * 32 + half * 8 + 16]);
                acc = __builtin_amdgcn_wmma_f32_16x16x32_f16(
                          false, A2[kc], false, B, (short)0, acc, false, false);
            }
            const float bias = sBe2[n];
            #pragma unroll
            for (int v = 0; v < 8; ++v) acc[v] = silu_f(acc[v] + bias);
            M[t] = acc;
        }
        // Stage edge_msg (f16) as A operand for the coord head.
        #pragma unroll
        for (int t = 0; t < 4; ++t) {
            const int n = lm + 16 * t;
            #pragma unroll
            for (int v = 0; v < 8; ++v)
                st[(v + 8 * half) * 64 + n] = (f16)M[t][v];
        }

        // ---- coord head: silu(msg @ Wc1 + bc1) @ Wc2 + bc2 ----
        v16h A3[2];
        #pragma unroll
        for (int kc = 0; kc < 2; ++kc)
            A3[kc] = make_frag(&st[lm * 64 + kc * 32 + half * 8],
                               &st[lm * 64 + kc * 32 + half * 8 + 16]);
        float part[8];
        #pragma unroll
        for (int v = 0; v < 8; ++v) part[v] = 0.0f;
        #pragma unroll
        for (int t = 0; t < 4; ++t) {
            v8f acc = {};
            const int n = lm + 16 * t;
            #pragma unroll
            for (int kc = 0; kc < 2; ++kc) {
                v16h B = make_frag(&sWTc[n * 64 + kc * 32 + half * 8],
                                   &sWTc[n * 64 + kc * 32 + half * 8 + 16]);
                acc = __builtin_amdgcn_wmma_f32_16x16x32_f16(
                          false, A3[kc], false, B, (short)0, acc, false, false);
            }
            const float bias = sBc1[n];
            const float w2   = sWc2v[n];
            #pragma unroll
            for (int v = 0; v < 8; ++v)
                part[v] += silu_f(acc[v] + bias) * w2;
        }
        // Reduce the n-dimension across the 16 lanes of each half-wave.
        #pragma unroll
        for (int off = 1; off < 16; off <<= 1) {
            #pragma unroll
            for (int v = 0; v < 8; ++v)
                part[v] += __shfl_xor(part[v], off, 32);
        }
        if (lm == 0) {
            #pragma unroll
            for (int v = 0; v < 8; ++v)
                sCw[wid][v + 8 * half] = part[v] + bc2v;  // edge m = v + 8*half
        }

        // ---- scatter: coord update (keyed on row) ----
        if (half == 0) {
            const int   rdst = sRow[wid][lm];
            const float cw   = sCw[wid][lm];
            atomicAdd(&posOut[rdst * 3 + 0], sRel[wid][lm][0] * cw);
            atomicAdd(&posOut[rdst * 3 + 1], sRel[wid][lm][1] * cw);
            atomicAdd(&posOut[rdst * 3 + 2], sRel[wid][lm][2] * cw);
        }
        // ---- scatter: message aggregation ----
        #pragma unroll
        for (int t = 0; t < 4; ++t) {
            const int n = lm + 16 * t;
            #pragma unroll
            for (int v = 0; v < 8; ++v) {
                const int m2   = v + 8 * half;
                const int rdst = sRow[wid][m2];
                atomicAdd(&msgAgg[rdst * 64 + n], M[t][v]);
            }
        }
    }
}

// ---------------------------------------------------------------------------
// Kernel 2: node_mlp over [h, msg_agg] (K=128). One wave = 16 nodes.
// ---------------------------------------------------------------------------
template <bool F16H>
__global__ __launch_bounds__(BLOCK_THREADS)
void egnn_node_kernel(const float* __restrict__ h,
                      const f16* __restrict__ h16,
                      const float* __restrict__ msgAgg,
                      const float* __restrict__ Wn1, const float* __restrict__ bn1,
                      const float* __restrict__ Wn2, const float* __restrict__ bn2,
                      float* __restrict__ hOut,
                      int N)
{
    __shared__ __align__(16) f16 sWT1[64 * 128];
    __shared__ __align__(16) f16 sWT2[64 * 64];
    __shared__ float sB1[64], sB2[64];
    __shared__ __align__(16) f16 sStage[WAVES_PER_BLOCK][16 * 128];

    const int tid = threadIdx.x;
    for (int i = tid; i < 128 * 64; i += BLOCK_THREADS) {
        int k = i >> 6, n = i & 63;
        sWT1[n * 128 + k] = (f16)Wn1[i];
    }
    for (int i = tid; i < 64 * 64; i += BLOCK_THREADS) {
        int k = i >> 6, n = i & 63;
        sWT2[n * 64 + k] = (f16)Wn2[i];
    }
    if (tid < 64) { sB1[tid] = bn1[tid]; sB2[tid] = bn2[tid]; }
    __syncthreads();

    const int lane = tid & 31, wid = tid >> 5;
    const int half = lane >> 4, lm = lane & 15;
    const int ntiles = N >> 4;   // N = 50000 is a multiple of 16
    f16* st = sStage[wid];

    for (int tile = blockIdx.x * WAVES_PER_BLOCK + wid; tile < ntiles;
         tile += gridDim.x * WAVES_PER_BLOCK)
    {
        const long long node = (long long)(tile << 4) + lm;
        f16* dst = st + lm * 128 + half * 64;
        if (F16H && half == 0) {
            // h side: already f16, straight 128B copy
            const uint4* sp = (const uint4*)(h16 + node * 64);
            uint4* d4 = (uint4*)dst;
            #pragma unroll
            for (int k4 = 0; k4 < 8; ++k4) d4[k4] = sp[k4];
        } else {
            const float4* src = (const float4*)((half ? msgAgg : h) + node * 64);
            #pragma unroll
            for (int k4 = 0; k4 < 16; ++k4) {
                float4 v = src[k4];
                dst[k4 * 4 + 0] = (f16)v.x;
                dst[k4 * 4 + 1] = (f16)v.y;
                dst[k4 * 4 + 2] = (f16)v.z;
                dst[k4 * 4 + 3] = (f16)v.w;
            }
        }

        // layer 1: [16 x 128] @ [128 x 64]
        v16h A[4];
        #pragma unroll
        for (int kc = 0; kc < 4; ++kc)
            A[kc] = make_frag(&st[lm * 128 + kc * 32 + half * 8],
                              &st[lm * 128 + kc * 32 + half * 8 + 16]);
        v8f C1[4];
        #pragma unroll
        for (int t = 0; t < 4; ++t) {
            v8f acc = {};
            const int n = lm + 16 * t;
            #pragma unroll
            for (int kc = 0; kc < 4; ++kc) {
                v16h B = make_frag(&sWT1[n * 128 + kc * 32 + half * 8],
                                   &sWT1[n * 128 + kc * 32 + half * 8 + 16]);
                acc = __builtin_amdgcn_wmma_f32_16x16x32_f16(
                          false, A[kc], false, B, (short)0, acc, false, false);
            }
            C1[t] = acc;
        }
        #pragma unroll
        for (int t = 0; t < 4; ++t) {
            const int n = lm + 16 * t;
            const float bias = sB1[n];
            #pragma unroll
            for (int v = 0; v < 8; ++v)
                st[(v + 8 * half) * 64 + n] = (f16)silu_f(C1[t][v] + bias);
        }

        // layer 2: [16 x 64] @ [64 x 64], write h_out fp32
        v16h A2[2];
        #pragma unroll
        for (int kc = 0; kc < 2; ++kc)
            A2[kc] = make_frag(&st[lm * 64 + kc * 32 + half * 8],
                               &st[lm * 64 + kc * 32 + half * 8 + 16]);
        #pragma unroll
        for (int t = 0; t < 4; ++t) {
            v8f acc = {};
            const int n = lm + 16 * t;
            #pragma unroll
            for (int kc = 0; kc < 2; ++kc) {
                v16h B = make_frag(&sWT2[n * 64 + kc * 32 + half * 8],
                                   &sWT2[n * 64 + kc * 32 + half * 8 + 16]);
                acc = __builtin_amdgcn_wmma_f32_16x16x32_f16(
                          false, A2[kc], false, B, (short)0, acc, false, false);
            }
            const float b2 = sB2[n];
            #pragma unroll
            for (int v = 0; v < 8; ++v) {
                const long long m2 = (long long)(tile << 4) + v + 8 * half;
                hOut[m2 * 64 + n] = acc[v] + b2;
            }
        }
    }
}

// ---------------------------------------------------------------------------
extern "C" void kernel_launch(void* const* d_in, const int* in_sizes, int n_in,
                              void* d_out, int out_size, void* d_ws, size_t ws_size,
                              hipStream_t stream)
{
    const float*     h    = (const float*)d_in[0];
    const float*     pos  = (const float*)d_in[1];
    const long long* eidx = (const long long*)d_in[2];
    const float* We1 = (const float*)d_in[3];
    const float* be1 = (const float*)d_in[4];
    const float* We2 = (const float*)d_in[5];
    const float* be2 = (const float*)d_in[6];
    const float* Wc1 = (const float*)d_in[7];
    const float* bc1 = (const float*)d_in[8];
    const float* Wc2 = (const float*)d_in[9];
    const float* bc2 = (const float*)d_in[10];
    const float* Wn1 = (const float*)d_in[11];
    const float* bn1 = (const float*)d_in[12];
    const float* Wn2 = (const float*)d_in[13];
    const float* bn2 = (const float*)d_in[14];

    const int N = in_sizes[0] / 64;
    const int E = in_sizes[2] / 2;

    float* hOut   = (float*)d_out;                     // [N,64]
    float* posOut = (float*)d_out + (long long)N * 64; // [N,3]

    // Workspace layout: [ msgAgg fp32 N*64 | h16 f16 N*64 (optional) ]
    const size_t msgBytes = (size_t)N * 64 * sizeof(float);
    const size_t needF16  = msgBytes + (size_t)N * 64 * sizeof(f16);
    float* msgAgg = (float*)d_ws;
    const bool useF16 = (ws_size >= needF16);
    f16* h16 = useF16 ? (f16*)((char*)d_ws + msgBytes) : nullptr;

    {   // init accumulators (+ one-time h -> f16 conversion)
        int total  = N * 64;
        int blocks = (total + 255) / 256;
        egnn_init_kernel<<<blocks, 256, 0, stream>>>(h, pos, msgAgg, posOut, h16, N);
    }
    if (useF16) {
        egnn_edge_kernel<true><<<1024, BLOCK_THREADS, 0, stream>>>(
            h, h16, pos, eidx, We1, be1, We2, be2, Wc1, bc1, Wc2, bc2,
            msgAgg, posOut, N, E);
        int blocks = (N / 16 + WAVES_PER_BLOCK - 1) / WAVES_PER_BLOCK;
        egnn_node_kernel<true><<<blocks, BLOCK_THREADS, 0, stream>>>(
            h, h16, msgAgg, Wn1, bn1, Wn2, bn2, hOut, N);
    } else {
        egnn_edge_kernel<false><<<1024, BLOCK_THREADS, 0, stream>>>(
            h, nullptr, pos, eidx, We1, be1, We2, be2, Wc1, bc1, Wc2, bc2,
            msgAgg, posOut, N, E);
        int blocks = (N / 16 + WAVES_PER_BLOCK - 1) / WAVES_PER_BLOCK;
        egnn_node_kernel<false><<<blocks, BLOCK_THREADS, 0, stream>>>(
            h, nullptr, msgAgg, Wn1, bn1, Wn2, bn2, hOut, N);
    }
}
